// MultiHeadAttention_81836306858028
// MI455X (gfx1250) — matrix operand order repaired and checked by reading, not compiled, not run on hardware
//
#include <hip/hip_runtime.h>

typedef __attribute__((ext_vector_type(16))) _Float16 v16h;
typedef __attribute__((ext_vector_type(8)))  _Float16 v8h;
typedef __attribute__((ext_vector_type(8)))  float    v8f;

#define DM   1024   // d_model
#define SL   2048   // seq len
#define NH   16
#define DH   64
#define BH   32     // B * NH

// ---- WMMA helpers -------------------------------------------------------
__device__ __forceinline__ v8f wmma_f16(v16h a, v16h b, v8f c) {
  // (neg_a, A, neg_b, B, c_mod, C, reuse_a, reuse_b)
  return __builtin_amdgcn_wmma_f32_16x16x32_f16(false, a, false, b, (short)0, c,
                                                false, false);
}

// A fragment (16x32 f16, row per lane): lane (hf, m): element e -> K = hf*16+e.
// One contiguous 16-half run, loaded as two 8-half halves.
__device__ __forceinline__ v16h load_a_frag(const _Float16* row, int kb, int hf) {
  v8h lo = *(const v8h*)(row + kb + hf * 16);
  v8h hi = *(const v8h*)(row + kb + hf * 16 + 8);
  return __builtin_shufflevector(lo, hi, 0, 1, 2, 3, 4, 5, 6, 7,
                                 8, 9, 10, 11, 12, 13, 14, 15);
}

// ---- conversion kernels -------------------------------------------------
__global__ void convert_f32_f16(const float* __restrict__ src,
                                _Float16* __restrict__ dst, int n) {
  int i = blockIdx.x * blockDim.x + threadIdx.x;
  if (i < n) dst[i] = (_Float16)src[i];
}

// W stored [in=1024][out=1024]; produce Wt f16 [out][in] so B fragments are
// contiguous-in-K per lane.
__global__ void transpose_f32_f16(const float* __restrict__ src,
                                  _Float16* __restrict__ dst) {
  int i = blockIdx.x * blockDim.x + threadIdx.x;   // over 1M elements
  int k = i & (DM - 1);
  int n = i >> 10;
  dst[(size_t)n * DM + k] = (_Float16)src[(size_t)k * DM + n];
}

// ---- GEMM: Y[4096x1024] = A[4096x1024] @ Bt^T + bias --------------------
// mode 0: store Q f16 [bh][s][d], scaled by 0.125 (1/sqrt(DH))
// mode 1: store K f16 [bh][s][d]
// mode 2: store V f16 transposed [bh][d][s]
// mode 3: store f32 [row][col]  (final output projection)
__global__ __launch_bounds__(256) void gemm_kernel(
    const _Float16* __restrict__ A, const _Float16* __restrict__ Bt,
    const float* __restrict__ bias, void* __restrict__ dst, int mode) {
  const int lane = threadIdx.x & 31;
  const int wave = threadIdx.x >> 5;
  const int hf   = lane >> 4;
  const int ln16 = lane & 15;
  const int mtile = blockIdx.x * 8 + wave;   // 0..255
  const int ngrp  = blockIdx.y;              // 0..15 (64 cols each)
  const int row0  = mtile * 16;

  v8f acc[4] = {};
  const _Float16* arow = A + (size_t)(row0 + ln16) * DM;

  for (int kb = 0; kb < DM; kb += 32) {
    v16h a = load_a_frag(arow, kb, hf);
#pragma unroll
    for (int g = 0; g < 4; ++g) {
      const int col = ngrp * 64 + g * 16 + ln16;
      v16h b = *(const v16h*)(Bt + (size_t)col * DM + kb + hf * 16);
      acc[g] = wmma_f16(a, b, acc[g]);
    }
  }

#pragma unroll
  for (int g = 0; g < 4; ++g) {
    const int col = ngrp * 64 + g * 16 + ln16;
    const float bc = bias[col];
#pragma unroll
    for (int r = 0; r < 8; ++r) {
      const int m    = hf ? 8 + r : r;
      const int srow = row0 + m;
      const float v  = acc[g][r] + bc;
      if (mode == 3) {
        ((float*)dst)[(size_t)srow * DM + col] = v;
      } else {
        const int bb = srow >> 11, s = srow & (SL - 1);
        const int h  = col >> 6,  d = col & (DH - 1);
        _Float16* o = (_Float16*)dst;
        if (mode == 0)
          o[(((size_t)(bb * NH + h)) * SL + s) * DH + d] = (_Float16)(v * 0.125f);
        else if (mode == 1)
          o[(((size_t)(bb * NH + h)) * SL + s) * DH + d] = (_Float16)v;
        else
          o[(((size_t)(bb * NH + h)) * DH + d) * SL + s] = (_Float16)v;
      }
    }
  }
}

// ---- Flash attention: per wave 16 query rows, stream 32-key blocks ------
__global__ __launch_bounds__(256) void attn_kernel(
    const _Float16* __restrict__ Qh, const _Float16* __restrict__ Kh,
    const _Float16* __restrict__ Vt, _Float16* __restrict__ ctx) {
  __shared__ _Float16 ps[8][16][32];   // per-wave P staging (C-layout -> A-layout)
  const int lane = threadIdx.x & 31;
  const int wave = threadIdx.x >> 5;
  const int hf   = lane >> 4;
  const int ln16 = lane & 15;
  const int bh    = blockIdx.y;                    // 0..31
  const int qbase = blockIdx.x * 128 + wave * 16;  // wave's 16 query rows

  const _Float16* qrow  = Qh + ((size_t)bh * SL + qbase + ln16) * DH;
  const v16h aq0 = load_a_frag(qrow, 0, hf);    // d 0..31  (pre-scaled by 1/8)
  const v16h aq1 = load_a_frag(qrow, 32, hf);   // d 32..63
  const _Float16* kbase = Kh + (size_t)bh * SL * DH;
  const _Float16* vbase = Vt + (size_t)bh * DH * SL;

  v8f acc[4] = {};
  float mrow[8], lrow[8];
#pragma unroll
  for (int r = 0; r < 8; ++r) { mrow[r] = -1e30f; lrow[r] = 0.f; }

  const int nblocks = ((qbase + 15) >> 5) + 1;   // causal key range
  for (int jb = 0; jb < nblocks; ++jb) {
    const int k0 = jb * 32;
    // ---- scores S = Q @ K^T : two 16x16 col-groups, K-dim = d (64 = 2x32)
    const _Float16* kr0 = kbase + (size_t)(k0 + ln16) * DH;
    const _Float16* kr1 = kbase + (size_t)(k0 + 16 + ln16) * DH;
    v16h bk00 = *(const v16h*)(kr0 + hf * 16);
    v16h bk01 = *(const v16h*)(kr0 + 32 + hf * 16);
    v16h bk10 = *(const v16h*)(kr1 + hf * 16);
    v16h bk11 = *(const v16h*)(kr1 + 32 + hf * 16);
    v8f s0 = {}; v8f s1 = {};
    s0 = wmma_f16(aq0, bk00, s0); s0 = wmma_f16(aq1, bk01, s0);
    s1 = wmma_f16(aq0, bk10, s1); s1 = wmma_f16(aq1, bk11, s1);

    if (k0 + 31 > qbase) {         // diagonal block: apply causal mask
#pragma unroll
      for (int r = 0; r < 8; ++r) {
        const int q = qbase + (hf ? 8 + r : r);
        if (k0 + ln16 > q)      s0[r] = -1e30f;
        if (k0 + 16 + ln16 > q) s1[r] = -1e30f;
      }
    }

    // ---- online softmax (row reductions across 16-lane halves)
    float corr[8];
#pragma unroll
    for (int r = 0; r < 8; ++r) {
      float t = fmaxf(s0[r], s1[r]);
      t = fmaxf(t, __shfl_xor(t, 1, 32));
      t = fmaxf(t, __shfl_xor(t, 2, 32));
      t = fmaxf(t, __shfl_xor(t, 4, 32));
      t = fmaxf(t, __shfl_xor(t, 8, 32));
      const float nm = fmaxf(mrow[r], t);
      corr[r] = __expf(mrow[r] - nm);
      const float p0 = __expf(s0[r] - nm);
      const float p1 = __expf(s1[r] - nm);
      s0[r] = p0; s1[r] = p1;
      float rs = p0 + p1;
      rs += __shfl_xor(rs, 1, 32);
      rs += __shfl_xor(rs, 2, 32);
      rs += __shfl_xor(rs, 4, 32);
      rs += __shfl_xor(rs, 8, 32);
      lrow[r] = lrow[r] * corr[r] + rs;
      mrow[r] = nm;
    }
#pragma unroll
    for (int g = 0; g < 4; ++g)
#pragma unroll
      for (int r = 0; r < 8; ++r) acc[g][r] *= corr[r];

    // ---- P: C-layout -> A-layout via per-wave LDS staging
#pragma unroll
    for (int r = 0; r < 8; ++r) {
      const int m = hf ? 8 + r : r;
      ps[wave][m][ln16]      = (_Float16)s0[r];
      ps[wave][m][16 + ln16] = (_Float16)s1[r];
    }
    asm volatile("s_wait_dscnt 0" ::: "memory");
    v8h plo = *(const v8h*)&ps[wave][ln16][hf * 16];
    v8h phi = *(const v8h*)&ps[wave][ln16][hf * 16 + 8];
    v16h pa = __builtin_shufflevector(plo, phi, 0, 1, 2, 3, 4, 5, 6, 7,
                                      8, 9, 10, 11, 12, 13, 14, 15);

    // ---- ctx += P(16x32) @ V(32x64); V transposed so K-dim contiguous
#pragma unroll
    for (int g = 0; g < 4; ++g) {
      v16h bv = *(const v16h*)(vbase + (size_t)(g * 16 + ln16) * SL + k0 + hf * 16);
      acc[g] = wmma_f16(pa, bv, acc[g]);
    }
  }

  // ---- epilogue: normalize and store ctx as f16 [b][s][h*64+d]
  const int bb = bh >> 4, h = bh & (NH - 1);
#pragma unroll
  for (int g = 0; g < 4; ++g) {
#pragma unroll
    for (int r = 0; r < 8; ++r) {
      const int m = hf ? 8 + r : r;
      const int s = qbase + m;
      const int col = h * DH + g * 16 + ln16;
      ctx[((size_t)(bb * SL + s)) * DM + col] = (_Float16)(acc[g][r] / lrow[r]);
    }
  }
}

// ---- host side ----------------------------------------------------------
extern "C" void kernel_launch(void* const* d_in, const int* in_sizes, int n_in,
                              void* d_out, int out_size, void* d_ws, size_t ws_size,
                              hipStream_t stream) {
  const float* x  = (const float*)d_in[0];
  // d_in[1] = mask (causal, implied in kernel)
  const float* Wq = (const float*)d_in[2];
  const float* bq = (const float*)d_in[3];
  const float* Wk = (const float*)d_in[4];
  const float* bk = (const float*)d_in[5];
  const float* Wv = (const float*)d_in[6];
  const float* bv = (const float*)d_in[7];
  const float* Wo = (const float*)d_in[8];
  const float* bo = (const float*)d_in[9];

  char* ws = (char*)d_ws;
  size_t off = 0;
  auto carve = [&](size_t bytes) -> void* {
    void* p = ws + off;
    off += (bytes + 255) & ~(size_t)255;
    return p;
  };
  const size_t NTOK = (size_t)2 * SL;           // 4096 rows
  _Float16* xh   = (_Float16*)carve(NTOK * DM * 2);
  _Float16* WqT  = (_Float16*)carve((size_t)DM * DM * 2);
  _Float16* WkT  = (_Float16*)carve((size_t)DM * DM * 2);
  _Float16* WvT  = (_Float16*)carve((size_t)DM * DM * 2);
  _Float16* WoT  = (_Float16*)carve((size_t)DM * DM * 2);
  _Float16* Qh   = (_Float16*)carve((size_t)BH * SL * DH * 2);
  _Float16* Kh   = (_Float16*)carve((size_t)BH * SL * DH * 2);
  _Float16* Vt   = (_Float16*)carve((size_t)BH * DH * SL * 2);
  _Float16* ctxh = (_Float16*)carve(NTOK * DM * 2);

  const int nx = (int)(NTOK * DM);
  convert_f32_f16<<<(nx + 255) / 256, 256, 0, stream>>>(x, xh, nx);
  transpose_f32_f16<<<(DM * DM) / 256, 256, 0, stream>>>(Wq, WqT);
  transpose_f32_f16<<<(DM * DM) / 256, 256, 0, stream>>>(Wk, WkT);
  transpose_f32_f16<<<(DM * DM) / 256, 256, 0, stream>>>(Wv, WvT);
  transpose_f32_f16<<<(DM * DM) / 256, 256, 0, stream>>>(Wo, WoT);

  dim3 gg(32, 16);  // 256 M-tiles (8 waves/block) x 16 col-groups of 64
  gemm_kernel<<<gg, 256, 0, stream>>>(xh, WqT, bq, Qh, 0);
  gemm_kernel<<<gg, 256, 0, stream>>>(xh, WkT, bk, Kh, 1);
  gemm_kernel<<<gg, 256, 0, stream>>>(xh, WvT, bv, Vt, 2);

  attn_kernel<<<dim3(16, BH), 256, 0, stream>>>(Qh, Kh, Vt, ctxh);

  gemm_kernel<<<gg, 256, 0, stream>>>(ctxh, WoT, bo, d_out, 3);
}
